// LlamaAttention_32066225832150
// MI455X (gfx1250) — compile-verified, matrix operand order checked by reading, not run on hardware
//
#include <hip/hip_runtime.h>

typedef __bf16 bf16;
typedef __attribute__((ext_vector_type(16))) __bf16 v16bf;
typedef __attribute__((ext_vector_type(8)))  float  v8f;

#define S_LEN 1024
#define DMODEL 4096
#define NH 32
#define NKV 8
#define HD 128
#define TCTX 2048      // CACHE + S
#define FDIM 4096      // NH*HD

// ---------- helpers ----------
__device__ __forceinline__ bf16 f2bf(float x) {
    unsigned u = __builtin_bit_cast(unsigned, x);
    u += 0x7FFFu + ((u >> 16) & 1u);          // round-to-nearest-even
    unsigned short h = (unsigned short)(u >> 16);
    return __builtin_bit_cast(bf16, h);
}

// Load a 16x16x32 bf16 A/B fragment row slice. `p` points at (row, k0) of a
// row-major matrix (K contiguous). Per ISA 16-bit A layout: lanes 0-15 take
// K pairs {half+2i, half+2i+1} with half=0, lanes 16-31 with half=8; vgprs
// 4..7 repeat at K+16. Compiler coalesces these into global_load_b128 pairs.
__device__ __forceinline__ v16bf load_frag(const bf16* p, int lane) {
    const int half = (lane & 16) ? 8 : 0;
    v16bf f;
#pragma unroll
    for (int i = 0; i < 4; ++i) {
        f[2*i]     = p[half + 2*i];
        f[2*i+1]   = p[half + 2*i + 1];
        f[8+2*i]   = p[16 + half + 2*i];
        f[8+2*i+1] = p[16 + half + 2*i + 1];
    }
    return f;
}

// ---------- elementwise kernels ----------
__global__ void cvt_f32_bf16(const float* __restrict__ src, bf16* __restrict__ dst, int n) {
    int i = blockIdx.x * blockDim.x + threadIdx.x;
    if (i < n) dst[i] = f2bf(src[i]);
}

// k_cache [NKV][HD][1024] -> kh_out [NKV][HD][TCTX] (t<1024) and KhT bf16 [NKV][TCTX][HD]
__global__ void cache_k_kernel(const float* __restrict__ kc, float* __restrict__ kh,
                               bf16* __restrict__ kht) {
    int i = blockIdx.x * blockDim.x + threadIdx.x;      // n*HD*1024
    if (i >= NKV * HD * 1024) return;
    int t = i % 1024, e = (i / 1024) % HD, n = i / (1024 * HD);
    float v = kc[i];
    kh[(n * HD + e) * TCTX + t] = v;
    kht[((size_t)n * TCTX + t) * HD + e] = f2bf(v);
}

// v_cache [NKV][1024][HD] -> vh_out [NKV][TCTX][HD] (t<1024) and VhT bf16 [NKV][HD][TCTX]
__global__ void cache_v_kernel(const float* __restrict__ vc, float* __restrict__ vh,
                               bf16* __restrict__ vht) {
    int i = blockIdx.x * blockDim.x + threadIdx.x;
    if (i >= NKV * 1024 * HD) return;
    int e = i % HD, t = (i / HD) % 1024, n = i / (1024 * HD);
    float v = vc[i];
    vh[((size_t)n * TCTX + t) * HD + e] = v;
    vht[((size_t)n * HD + e) * TCTX + t] = f2bf(v);
}

// q_pre [S][FDIM] f32 -> RoPE -> Qb bf16 [NH][S][HD]
__global__ void rope_q_kernel(const float* __restrict__ q, const float* __restrict__ cosp,
                              const float* __restrict__ sinp, bf16* __restrict__ qb) {
    int i = blockIdx.x * blockDim.x + threadIdx.x;      // s*NH*64
    if (i >= S_LEN * NH * (HD / 2)) return;
    int e = i % (HD / 2), h = (i / (HD / 2)) % NH, s = i / ((HD / 2) * NH);
    float xr = q[(size_t)s * FDIM + h * HD + e];
    float xi = q[(size_t)s * FDIM + h * HD + e + HD / 2];
    float c = cosp[s * (HD / 2) + e], sn = sinp[s * (HD / 2) + e];
    size_t base = ((size_t)h * S_LEN + s) * HD;
    qb[base + e]          = f2bf(xr * c - xi * sn);
    qb[base + e + HD / 2] = f2bf(xr * sn + xi * c);
}

// k_pre [S][NKV*HD] f32 -> RoPE -> kh_out f32 [NKV][HD][TCTX] (t=1024+s), KhT bf16 [NKV][TCTX][HD]
__global__ void rope_k_kernel(const float* __restrict__ k, const float* __restrict__ cosp,
                              const float* __restrict__ sinp, float* __restrict__ kh,
                              bf16* __restrict__ kht) {
    int i = blockIdx.x * blockDim.x + threadIdx.x;      // s*NKV*64
    if (i >= S_LEN * NKV * (HD / 2)) return;
    int e = i % (HD / 2), n = (i / (HD / 2)) % NKV, s = i / ((HD / 2) * NKV);
    float xr = k[(size_t)s * (NKV * HD) + n * HD + e];
    float xi = k[(size_t)s * (NKV * HD) + n * HD + e + HD / 2];
    float c = cosp[s * (HD / 2) + e], sn = sinp[s * (HD / 2) + e];
    float o0 = xr * c - xi * sn, o1 = xr * sn + xi * c;
    int t = 1024 + s;
    kh[((size_t)n * HD + e) * TCTX + t]            = o0;
    kh[((size_t)n * HD + e + HD / 2) * TCTX + t]   = o1;
    kht[((size_t)n * TCTX + t) * HD + e]           = f2bf(o0);
    kht[((size_t)n * TCTX + t) * HD + e + HD / 2]  = f2bf(o1);
}

// v_pre [S][NKV*HD] f32 -> vh_out f32 [NKV][TCTX][HD] (t=1024+s), VhT bf16 [NKV][HD][TCTX]
__global__ void v_epi_kernel(const float* __restrict__ v, float* __restrict__ vh,
                             bf16* __restrict__ vht) {
    int i = blockIdx.x * blockDim.x + threadIdx.x;      // s*NKV*HD
    if (i >= S_LEN * NKV * HD) return;
    int e = i % HD, n = (i / HD) % NKV, s = i / (HD * NKV);
    float x = v[(size_t)s * (NKV * HD) + n * HD + e];
    int t = 1024 + s;
    vh[((size_t)n * TCTX + t) * HD + e] = x;
    vht[((size_t)n * HD + e) * TCTX + t] = f2bf(x);
}

// ---------- register-blocked WMMA GEMM ----------
// C[M][N] = A[M][K] * W[N][K]^T, bf16 in, f32 out.
// One wave computes a 32x64 output block = 2x4 WMMA tiles: 8 v_wmma per
// 32-wide K step on 6 fragment loads. 8 accumulators (64 VGPRs) + fragments
// stay well under 256 VGPRs -> no scratch spills (4x4 spilled).
__global__ void __launch_bounds__(256, 1)
gemm_bf16_wmma(const bf16* __restrict__ A, const bf16* __restrict__ W,
               float* __restrict__ C, int M, int N, int K) {
    int gtid = blockIdx.x * blockDim.x + threadIdx.x;
    int wave = gtid >> 5;
    int lane = threadIdx.x & 31;
    int nblk = N >> 6;                          // 64-col blocks
    int blocks = (M >> 5) * nblk;               // 32-row blocks
    if (wave >= blocks) return;                 // wave-uniform exit
    int mb = wave / nblk, nb = wave % nblk;

    v8f acc[2][4];
#pragma unroll
    for (int i = 0; i < 2; ++i)
#pragma unroll
        for (int j = 0; j < 4; ++j)
#pragma unroll
            for (int r = 0; r < 8; ++r) acc[i][j][r] = 0.0f;

    const bf16* arow[2];
    const bf16* brow[4];
#pragma unroll
    for (int i = 0; i < 2; ++i)
        arow[i] = A + (size_t)(mb * 32 + i * 16 + (lane & 15)) * K;
#pragma unroll
    for (int j = 0; j < 4; ++j)
        brow[j] = W + (size_t)(nb * 64 + j * 16 + (lane & 15)) * K;

    for (int k0 = 0; k0 < K; k0 += 32) {
        v16bf a[2], b[4];
#pragma unroll
        for (int i = 0; i < 2; ++i) a[i] = load_frag(arow[i] + k0, lane);
#pragma unroll
        for (int j = 0; j < 4; ++j) b[j] = load_frag(brow[j] + k0, lane);
#pragma unroll
        for (int i = 0; i < 2; ++i)
#pragma unroll
            for (int j = 0; j < 4; ++j)
                acc[i][j] = __builtin_amdgcn_wmma_f32_16x16x32_bf16(
                    false, a[i], false, b[j], (short)0, acc[i][j], false, false);
    }

    int rhalf = (lane & 16) ? 8 : 0;
    int cl    = lane & 15;
#pragma unroll
    for (int i = 0; i < 2; ++i) {
#pragma unroll
        for (int r = 0; r < 8; ++r) {
            size_t row = (size_t)(mb * 32 + i * 16 + rhalf + r) * N + nb * 64 + cl;
#pragma unroll
            for (int j = 0; j < 4; ++j) C[row + j * 16] = acc[i][j][r];
        }
    }
}

// ---------- flash attention: one wave per (head, 16-query tile) ----------
__global__ void flash_attn_kernel(const bf16* __restrict__ Qb,   // [NH][S][HD]
                                  const bf16* __restrict__ KhT,  // [NKV][TCTX][HD]
                                  const bf16* __restrict__ VhT,  // [NKV][HD][TCTX]
                                  const float* __restrict__ mask,// [S][TCTX]
                                  bf16* __restrict__ Yb) {       // [S][FDIM]
    const float scale = 0.08838834764831845f;  // 1/sqrt(128)
    int wib  = threadIdx.x >> 5;
    int lane = threadIdx.x & 31;
    int gw = blockIdx.x * (blockDim.x >> 5) + wib;
    if (gw >= NH * (S_LEN / 16)) return;
    int head = gw / (S_LEN / 16);
    int mt   = gw % (S_LEN / 16);
    int kvh  = head / (NH / NKV);

    __shared__ bf16 plds[4][16][32];           // per-wave P staging tile
    bf16 (*pw)[32] = plds[wib];

    // Q fragments: 16 x 128 split into 4 K-chunks of 32
    v16bf qa[4];
    const bf16* qrow = Qb + ((size_t)head * S_LEN + mt * 16 + (lane & 15)) * HD;
#pragma unroll
    for (int c = 0; c < 4; ++c) qa[c] = load_frag(qrow + 32 * c, lane);

    v8f o[8];
#pragma unroll
    for (int e = 0; e < 8; ++e)
#pragma unroll
        for (int i = 0; i < 8; ++i) o[e][i] = 0.0f;
    float m_i[8], l_i[8];
#pragma unroll
    for (int r = 0; r < 8; ++r) { m_i[r] = -3.0e38f; l_i[r] = 0.0f; }

    const int col   = lane & 15;
    const int rbase = mt * 16 + ((lane & 16) ? 8 : 0);

    for (int t0 = 0; t0 < TCTX; t0 += 32) {
        // S = Q * K^T  (two 16-col key tiles)
        v8f s0, s1;
#pragma unroll
        for (int i = 0; i < 8; ++i) { s0[i] = 0.0f; s1[i] = 0.0f; }
        const bf16* kr0 = KhT + ((size_t)kvh * TCTX + t0 +      (lane & 15)) * HD;
        const bf16* kr1 = KhT + ((size_t)kvh * TCTX + t0 + 16 + (lane & 15)) * HD;
#pragma unroll
        for (int c = 0; c < 4; ++c) {
            v16bf b0 = load_frag(kr0 + 32 * c, lane);
            s0 = __builtin_amdgcn_wmma_f32_16x16x32_bf16(false, qa[c], false, b0,
                                                         (short)0, s0, false, false);
            v16bf b1 = load_frag(kr1 + 32 * c, lane);
            s1 = __builtin_amdgcn_wmma_f32_16x16x32_bf16(false, qa[c], false, b1,
                                                         (short)0, s1, false, false);
        }
        // online softmax per row (rows live in vgpr index; cols in lanes 0-15 / 16-31 halves)
#pragma unroll
        for (int r = 0; r < 8; ++r) {
            float a0 = s0[r] * scale + mask[(size_t)(rbase + r) * TCTX + t0 + col];
            float a1 = s1[r] * scale + mask[(size_t)(rbase + r) * TCTX + t0 + 16 + col];
            float mx = fmaxf(a0, a1);
#pragma unroll
            for (int off = 1; off < 16; off <<= 1) mx = fmaxf(mx, __shfl_xor(mx, off, 32));
            float mnew  = fmaxf(m_i[r], mx);
            float alpha = __expf(m_i[r] - mnew);
            m_i[r] = mnew;
            float p0 = __expf(a0 - mnew);
            float p1 = __expf(a1 - mnew);
            float ps = p0 + p1;
#pragma unroll
            for (int off = 1; off < 16; off <<= 1) ps += __shfl_xor(ps, off, 32);
            l_i[r] = l_i[r] * alpha + ps;
#pragma unroll
            for (int e = 0; e < 8; ++e) o[e][r] *= alpha;
            int prow = r + ((lane & 16) ? 8 : 0);
            pw[prow][col]      = f2bf(p0);
            pw[prow][col + 16] = f2bf(p1);
        }
        // reload P in A-fragment layout (wave-private LDS; DScnt ordering by compiler)
        v16bf pa = load_frag(&pw[lane & 15][0], lane);
        // O += P * V
        const bf16* vbase = VhT + (size_t)kvh * HD * TCTX + t0;
#pragma unroll
        for (int e = 0; e < 8; ++e) {
            v16bf vb = load_frag(vbase + (size_t)(e * 16 + (lane & 15)) * TCTX, lane);
            o[e] = __builtin_amdgcn_wmma_f32_16x16x32_bf16(false, pa, false, vb,
                                                           (short)0, o[e], false, false);
        }
    }
    // epilogue: normalize, write attention output (bf16 staging for final GEMM)
#pragma unroll
    for (int r = 0; r < 8; ++r) {
        float inv = 1.0f / l_i[r];
        size_t row = (size_t)(rbase + r) * FDIM + head * HD;
#pragma unroll
        for (int e = 0; e < 8; ++e)
            Yb[row + e * 16 + (lane & 15)] = f2bf(o[e][r] * inv);
    }
}

// ---------- host launch ----------
extern "C" void kernel_launch(void* const* d_in, const int* in_sizes, int n_in,
                              void* d_out, int out_size, void* d_ws, size_t ws_size,
                              hipStream_t stream) {
    const float* hidden = (const float*)d_in[0];
    const float* fcos   = (const float*)d_in[1];
    const float* fsin   = (const float*)d_in[2];
    const float* amask  = (const float*)d_in[3];
    const float* kcache = (const float*)d_in[4];
    const float* vcache = (const float*)d_in[5];
    const float* wq     = (const float*)d_in[6];
    const float* wk     = (const float*)d_in[7];
    const float* wv     = (const float*)d_in[8];
    const float* wo     = (const float*)d_in[9];

    float* y_out  = (float*)d_out;                                   // [S][D]
    float* kh_out = y_out + (size_t)S_LEN * DMODEL;                  // [NKV][HD][TCTX]
    float* vh_out = kh_out + (size_t)NKV * HD * TCTX;                // [NKV][TCTX][HD]

    // workspace carve-up
    char* ws = (char*)d_ws;
    size_t off = 0;
    auto carve = [&](size_t bytes) { char* p = ws + off; off += (bytes + 255) & ~(size_t)255; return p; };
    bf16* hs_b  = (bf16*)carve((size_t)S_LEN * DMODEL * 2);
    bf16* wq_b  = (bf16*)carve((size_t)FDIM * DMODEL * 2);
    bf16* wk_b  = (bf16*)carve((size_t)NKV * HD * DMODEL * 2);
    bf16* wv_b  = (bf16*)carve((size_t)NKV * HD * DMODEL * 2);
    bf16* wo_b  = (bf16*)carve((size_t)DMODEL * FDIM * 2);
    float* q_pre = (float*)carve((size_t)S_LEN * FDIM * 4);
    float* k_pre = (float*)carve((size_t)S_LEN * NKV * HD * 4);
    float* v_pre = (float*)carve((size_t)S_LEN * NKV * HD * 4);
    bf16* Qb  = (bf16*)carve((size_t)NH * S_LEN * HD * 2);
    bf16* KhT = (bf16*)carve((size_t)NKV * TCTX * HD * 2);
    bf16* VhT = (bf16*)carve((size_t)NKV * HD * TCTX * 2);
    bf16* Yb  = (bf16*)carve((size_t)S_LEN * FDIM * 2);

    auto blks = [](size_t n, int b) { return (unsigned)((n + b - 1) / b); };

    // 1) bf16 conversions
    cvt_f32_bf16<<<blks((size_t)S_LEN * DMODEL, 256), 256, 0, stream>>>(hidden, hs_b, S_LEN * DMODEL);
    cvt_f32_bf16<<<blks((size_t)FDIM * DMODEL, 256), 256, 0, stream>>>(wq, wq_b, FDIM * DMODEL);
    cvt_f32_bf16<<<blks((size_t)NKV * HD * DMODEL, 256), 256, 0, stream>>>(wk, wk_b, NKV * HD * DMODEL);
    cvt_f32_bf16<<<blks((size_t)NKV * HD * DMODEL, 256), 256, 0, stream>>>(wv, wv_b, NKV * HD * DMODEL);
    cvt_f32_bf16<<<blks((size_t)DMODEL * FDIM, 256), 256, 0, stream>>>(wo, wo_b, DMODEL * FDIM);

    // 2) cache copies into outputs + bf16 staging
    cache_k_kernel<<<blks((size_t)NKV * HD * 1024, 256), 256, 0, stream>>>(kcache, kh_out, KhT);
    cache_v_kernel<<<blks((size_t)NKV * 1024 * HD, 256), 256, 0, stream>>>(vcache, vh_out, VhT);

    // 3) QKV projections (one 32x64 block per wave, 8 waves / 256-thread block)
    {
        size_t waves_q = (S_LEN / 32) * (FDIM / 64);
        gemm_bf16_wmma<<<blks(waves_q * 32, 256), 256, 0, stream>>>(hs_b, wq_b, q_pre, S_LEN, FDIM, DMODEL);
        size_t waves_kv = (S_LEN / 32) * ((NKV * HD) / 64);
        gemm_bf16_wmma<<<blks(waves_kv * 32, 256), 256, 0, stream>>>(hs_b, wk_b, k_pre, S_LEN, NKV * HD, DMODEL);
        gemm_bf16_wmma<<<blks(waves_kv * 32, 256), 256, 0, stream>>>(hs_b, wv_b, v_pre, S_LEN, NKV * HD, DMODEL);
    }

    // 4) RoPE + V epilogues (also fill kh/vh outputs for t >= 1024)
    rope_q_kernel<<<blks((size_t)S_LEN * NH * (HD / 2), 256), 256, 0, stream>>>(q_pre, fcos, fsin, Qb);
    rope_k_kernel<<<blks((size_t)S_LEN * NKV * (HD / 2), 256), 256, 0, stream>>>(k_pre, fcos, fsin, kh_out, KhT);
    v_epi_kernel<<<blks((size_t)S_LEN * NKV * HD, 256), 256, 0, stream>>>(v_pre, vh_out, VhT);

    // 5) flash attention: 2048 waves, 4 waves / 128-thread block
    flash_attn_kernel<<<(NH * (S_LEN / 16)) / 4, 128, 0, stream>>>(Qb, KhT, VhT, amask, Yb);

    // 6) output projection into y
    {
        size_t waves_o = (S_LEN / 32) * (DMODEL / 64);
        gemm_bf16_wmma<<<blks(waves_o * 32, 256), 256, 0, stream>>>(Yb, wo_b, y_out, S_LEN, DMODEL, FDIM);
    }
}
